// MultiHeadSelfAttention_24867860644106
// MI455X (gfx1250) — matrix operand order repaired and checked by reading, not compiled, not run on hardware
//
#include <hip/hip_runtime.h>
#include <math.h>

// ---- problem constants (fixed by the reference) ----
#define B_   2
#define S_   2048
#define H_   2048
#define NH_  16
#define HD_  128
#define M_   (B_ * S_)   // 4096 rows of the flattened [B*S, H] activations

typedef _Float16 f16;
typedef __attribute__((ext_vector_type(8)))  _Float16 v8h;
typedef __attribute__((ext_vector_type(16))) _Float16 v16h;
typedef __attribute__((ext_vector_type(8)))  float    v8f;
typedef __attribute__((ext_vector_type(4)))  int      v4i;

// async global->LDS path (gfx1250). Guarded: falls back to load+ds_store if the
// toolchain doesn't declare the builtins. Histogram shows which path lowered.
#if defined(__has_builtin)
#  if __has_builtin(__builtin_amdgcn_global_load_async_to_lds_b128) && \
      __has_builtin(__builtin_amdgcn_s_wait_asynccnt)
#    define USE_ASYNC_LDS 1
#  endif
#endif
#ifndef USE_ASYNC_LDS
#  define USE_ASYNC_LDS 0
#endif

#if USE_ASYNC_LDS
// builtin signature (from compiler diagnostics):
//   (as1 v4i* src, as3 v4i* dst, imm offset, imm cpol)
#define ASYNC_B128(srcp, dstp, OFF)                                     \
    __builtin_amdgcn_global_load_async_to_lds_b128(                     \
        (__attribute__((address_space(1))) v4i*)(v4i*)(srcp),           \
        (__attribute__((address_space(3))) v4i*)(v4i*)(dstp), OFF, 0)
#endif

static __device__ inline v16h cat8(v8h a, v8h b) {
    return __builtin_shufflevector(a, b, 0,1,2,3,4,5,6,7,8,9,10,11,12,13,14,15);
}

// ---------------------------------------------------------------------------
// fp32 -> f16 conversion (x and the four weight matrices)
// ---------------------------------------------------------------------------
__global__ __launch_bounds__(256) void cvt_kernel(const float* __restrict__ in,
                                                  f16* __restrict__ out, int n) {
    int i = blockIdx.x * blockDim.x + threadIdx.x;
    int stride = gridDim.x * blockDim.x;
    for (; i < n; i += stride) out[i] = (f16)in[i];
}

// ---------------------------------------------------------------------------
// GEMM: out[m,n] = scale * (sum_k A[m,k] * W[n,k] + bias[n])
//   A: f16 [M_, H_] row-major, W: f16 [H_, H_] row-major (torch Linear weight)
//   MODE 0: f16 out, row-major [M_, H_]          (Q, K projections)
//   MODE 1: f16 out, transposed to [B,NH,HD,S]   (V projection -> V^T)
//   MODE 2: f32 out, row-major [M_, H_]          (output projection)
// Each wave computes a 32(M) x 64(N) tile: 2 A-frags x 4 B-frags -> 8
// independent WMMAs per K-step. Block = 8 waves = 64(M) x 256(N).
// Grid = (M_/64, H_/256).
// ---------------------------------------------------------------------------
template<int MODE>
__global__ __launch_bounds__(256) void gemm_xwT(const f16* __restrict__ A,
                                                const f16* __restrict__ W,
                                                const float* __restrict__ bias,
                                                float scale,
                                                void* __restrict__ outp) {
    constexpr int K = H_;
    constexpr int N = H_;
    const int wave = threadIdx.x >> 5;
    const int lane = threadIdx.x & 31;
    const int l16  = lane & 15;
    const int hi   = lane >> 4;

    const int mBase = blockIdx.x * 64  + (wave & 1) * 32;
    const int nBase = blockIdx.y * 256 + (wave >> 1) * 64;

    v8f acc[2][4] = {};

    // A fragment: lane half owns K-subchunks {base, base+16}, 8 halves each
    const f16* aRow0 = A + (size_t)(mBase + l16) * K + (hi ? 8 : 0);
    const f16* aRow1 = aRow0 + (size_t)16 * K;
    // B fragment: lane half owns K-subchunks {base, base+16}, 8 halves each
    const f16* wRow0 = W + (size_t)(nBase + l16) * K + (hi ? 8 : 0);

    for (int k = 0; k < K; k += 32) {
        v16h af0 = cat8(*(const v8h*)(aRow0 + k), *(const v8h*)(aRow0 + k + 16));
        v16h af1 = cat8(*(const v8h*)(aRow1 + k), *(const v8h*)(aRow1 + k + 16));
#pragma unroll
        for (int j = 0; j < 4; ++j) {
            const f16* wr = wRow0 + (size_t)j * 16 * K + k;
            v16h bf = cat8(*(const v8h*)wr, *(const v8h*)(wr + 16));
            acc[0][j] = __builtin_amdgcn_wmma_f32_16x16x32_f16(
                false, af0, false, bf, (short)0, acc[0][j], false, false);
            acc[1][j] = __builtin_amdgcn_wmma_f32_16x16x32_f16(
                false, af1, false, bf, (short)0, acc[1][j], false, false);
        }
    }

#pragma unroll
    for (int i = 0; i < 2; ++i) {
#pragma unroll
        for (int j = 0; j < 4; ++j) {
            const int n = nBase + j * 16 + l16;
            const float bv = bias[n];
#pragma unroll
            for (int r = 0; r < 8; ++r) {
                const int m = mBase + i * 16 + r + hi * 8; // C layout: lanes 0-15 M=r, 16-31 M=8+r
                const float val = (acc[i][j][r] + bv) * scale;
                if constexpr (MODE == 0) {
                    ((f16*)outp)[(size_t)m * N + n] = (f16)val;
                } else if constexpr (MODE == 2) {
                    ((float*)outp)[(size_t)m * N + n] = val;
                } else {
                    // V^T layout: [B, NH, HD, S]
                    const int b  = m / S_, s = m % S_;
                    const int hh = n / HD_, d = n % HD_;
                    ((f16*)outp)[((size_t)((b * NH_ + hh) * HD_) + d) * S_ + s] = (f16)val;
                }
            }
        }
    }
}

// ---------------------------------------------------------------------------
// Flash attention. Block = 8 waves sharing one (b,h); each wave owns one
// 16-query tile (block covers 128 queries). Per 32-key step the block
// cooperatively stages the K tile (32x128) and V^T tile (128x32) into LDS
// (async global->LDS when available), then each wave runs:
//   8 score WMMAs -> online softmax (exp2 domain, cross-lane shfl reductions)
//   -> P transposed through per-wave LDS into A-layout -> 8 PV WMMAs.
//   Q: f16 [B,S,H] pre-scaled by log2(e)/sqrt(HD);  K: f16 [B,S,H]
//   Vt: f16 [B,NH,HD,S];  ctx out: f16 [B,S,H]
// ---------------------------------------------------------------------------
__global__ __launch_bounds__(256) void attn_kernel(const f16* __restrict__ Q,
                                                   const f16* __restrict__ Kx,
                                                   const f16* __restrict__ Vt,
                                                   f16* __restrict__ ctx) {
    __shared__ alignas(16) f16 kt[32][HD_];      // 8 KB: K tile, [key][d]
    __shared__ alignas(16) f16 vtile[HD_][32];   // 8 KB: V^T tile, [d][key]
    __shared__ alignas(16) f16 pbuf[8][16 * 32]; // 8 KB: per-wave P transpose

    const int tid  = threadIdx.x;
    const int wave = tid >> 5;
    const int lane = tid & 31;
    const int l16  = lane & 15;
    const int hi   = lane >> 4;

    const int bh   = blockIdx.x >> 4;     // 16 blocks per (b,h)
    const int qblk = blockIdx.x & 15;
    const int h    = bh & (NH_ - 1);
    const int b    = bh >> 4;
    const int qt   = qblk * 8 + wave;     // this wave's 16-query tile

    // Load the 16x128 Q tile as 4 A-fragments (K=32 each), kept in registers.
    const size_t qRow = (size_t)(b * S_ + qt * 16 + l16) * H_ + h * HD_;
    v16h qf[4];
#pragma unroll
    for (int c = 0; c < 4; ++c) {
        const f16* p = Q + qRow + c * 32 + (hi ? 8 : 0);
        qf[c] = cat8(*(const v8h*)p, *(const v8h*)(p + 16));
    }

    v8f acc[8] = {};
    float rmax[8], rsum[8];
#pragma unroll
    for (int r = 0; r < 8; ++r) { rmax[r] = -3.0e38f; rsum[r] = 0.0f; }

    // cooperative staging assignments (256 threads x 32 B per tile)
    const int kKey = tid >> 3, kD = (tid & 7) * 16;
    const f16* kSrc0 = Kx + (size_t)(b * S_ + kKey) * H_ + h * HD_ + kD;
    f16* kDst = &kt[kKey][kD];
    const int vD = tid >> 1, vK = (tid & 1) * 16;
    const f16* vSrc0 = Vt + ((size_t)((b * NH_ + h) * HD_) + vD) * S_ + vK;
    f16* vDst = &vtile[vD][vK];

    f16* myp = &pbuf[wave][0];

    for (int kb = 0; kb < S_; kb += 32) {
        __syncthreads();   // previous iteration's tile reads are done
        const f16* ks = kSrc0 + (size_t)kb * H_;
        const f16* vs = vSrc0 + kb;
#if USE_ASYNC_LDS
        ASYNC_B128(ks,     kDst,     0);
        ASYNC_B128(ks + 8, kDst + 8, 0);
        ASYNC_B128(vs,     vDst,     0);
        ASYNC_B128(vs + 8, vDst + 8, 0);
        __builtin_amdgcn_s_wait_asynccnt(0);
#else
        *(v16h*)kDst = *(const v16h*)ks;
        *(v16h*)vDst = *(const v16h*)vs;
#endif
        __syncthreads();   // tiles visible to all waves

        // ---- scores: two 16x16 tiles (keys kb..kb+15, kb+16..kb+31) ----
        v8f sc[2] = {};
#pragma unroll
        for (int t = 0; t < 2; ++t) {
#pragma unroll
            for (int c = 0; c < 4; ++c) {
                const f16* kp = &kt[t * 16 + l16][c * 32 + (hi ? 8 : 0)];
                v16h bf = cat8(*(const v8h*)kp, *(const v8h*)(kp + 16));
                sc[t] = __builtin_amdgcn_wmma_f32_16x16x32_f16(
                    false, qf[c], false, bf, (short)0, sc[t], false, false);
            }
        }

        // ---- online softmax (exp2 domain; Q carried the log2e/sqrt scale) ----
        float bm[8];
#pragma unroll
        for (int r = 0; r < 8; ++r) bm[r] = fmaxf(sc[0][r], sc[1][r]);
#pragma unroll
        for (int m = 1; m < 16; m <<= 1)
#pragma unroll
            for (int r = 0; r < 8; ++r)
                bm[r] = fmaxf(bm[r], __shfl_xor(bm[r], m, 32));

        float corr[8], ps[8];
#pragma unroll
        for (int r = 0; r < 8; ++r) {
            const float nm = fmaxf(rmax[r], bm[r]);
            corr[r] = exp2f(rmax[r] - nm);
            rmax[r] = nm;
            const float p0 = exp2f(sc[0][r] - nm);
            const float p1 = exp2f(sc[1][r] - nm);
            sc[0][r] = p0; sc[1][r] = p1;
            ps[r] = p0 + p1;
        }
#pragma unroll
        for (int m = 1; m < 16; m <<= 1)
#pragma unroll
            for (int r = 0; r < 8; ++r) ps[r] += __shfl_xor(ps[r], m, 32);
#pragma unroll
        for (int r = 0; r < 8; ++r) rsum[r] = rsum[r] * corr[r] + ps[r];
#pragma unroll
        for (int t = 0; t < 8; ++t)
#pragma unroll
            for (int r = 0; r < 8; ++r) acc[t][r] *= corr[r];

        // ---- transpose P (C layout -> A layout) through per-wave LDS ----
#pragma unroll
        for (int r = 0; r < 8; ++r) {
            const int row = r + hi * 8;
            myp[row * 32 + l16]      = (f16)sc[0][r];
            myp[row * 32 + 16 + l16] = (f16)sc[1][r];
        }
        const f16* pa = myp + l16 * 32 + (hi ? 8 : 0);
        v16h af = cat8(*(const v8h*)pa, *(const v8h*)(pa + 16));

        // ---- ctx += P @ V from the shared V^T tile ----
#pragma unroll
        for (int t = 0; t < 8; ++t) {
            const f16* vp = &vtile[t * 16 + l16][hi * 8];
            v16h bf = cat8(*(const v8h*)vp, *(const v8h*)(vp + 16));
            acc[t] = __builtin_amdgcn_wmma_f32_16x16x32_f16(
                false, af, false, bf, (short)0, acc[t], false, false);
        }
    }

    // ---- normalize and store ctx in [B,S,H] (f16) ----
    float inv[8];
#pragma unroll
    for (int r = 0; r < 8; ++r) inv[r] = 1.0f / rsum[r];
#pragma unroll
    for (int t = 0; t < 8; ++t) {
#pragma unroll
        for (int r = 0; r < 8; ++r) {
            const int srow = qt * 16 + r + hi * 8;
            ctx[(size_t)(b * S_ + srow) * H_ + h * HD_ + t * 16 + l16] =
                (f16)(acc[t][r] * inv[r]);
        }
    }
}

// ---------------------------------------------------------------------------
// Host-side launcher
// ---------------------------------------------------------------------------
extern "C" void kernel_launch(void* const* d_in, const int* in_sizes, int n_in,
                              void* d_out, int out_size, void* d_ws, size_t ws_size,
                              hipStream_t stream) {
    const float* x  = (const float*)d_in[0];
    const float* Wq = (const float*)d_in[1];
    const float* bq = (const float*)d_in[2];
    const float* Wk = (const float*)d_in[3];
    const float* bk = (const float*)d_in[4];
    const float* Wv = (const float*)d_in[5];
    const float* bv = (const float*)d_in[6];
    const float* Wo = (const float*)d_in[7];
    const float* bo = (const float*)d_in[8];

    const size_t XE = (size_t)M_ * H_;   // 8,388,608 elements
    const size_t WE = (size_t)H_ * H_;   // 4,194,304 elements

    // workspace layout (f16): x | Wq | Wk | Wv | Wo | Q | K | V^T | ctx  (~112 MB)
    f16* xh  = (f16*)d_ws;
    f16* wqh = xh  + XE;
    f16* wkh = wqh + WE;
    f16* wvh = wkh + WE;
    f16* woh = wvh + WE;
    f16* qh  = woh + WE;
    f16* kh  = qh  + XE;
    f16* vt  = kh  + XE;
    f16* ch  = vt  + XE;

    cvt_kernel<<<2048, 256, 0, stream>>>(x,  xh,  (int)XE);
    cvt_kernel<<<1024, 256, 0, stream>>>(Wq, wqh, (int)WE);
    cvt_kernel<<<1024, 256, 0, stream>>>(Wk, wkh, (int)WE);
    cvt_kernel<<<1024, 256, 0, stream>>>(Wv, wvh, (int)WE);
    cvt_kernel<<<1024, 256, 0, stream>>>(Wo, woh, (int)WE);

    const dim3 ggrid(M_ / 64, H_ / 256);   // (64, 8)
    // fold softmax 1/sqrt(HD) and the exp->exp2 log2(e) factor into Q
    const float qscale = 1.4426950408889634f / sqrtf((float)HD_);

    gemm_xwT<0><<<ggrid, 256, 0, stream>>>(xh, wqh, bq, qscale, qh);
    gemm_xwT<0><<<ggrid, 256, 0, stream>>>(xh, wkh, bk, 1.0f,   kh);
    gemm_xwT<1><<<ggrid, 256, 0, stream>>>(xh, wvh, bv, 1.0f,   vt);

    attn_kernel<<<(B_ * NH_ * (S_ / 16)) / 8, 256, 0, stream>>>(qh, kh, vt, ch);

    gemm_xwT<2><<<ggrid, 256, 0, stream>>>(ch, woh, bo, 1.0f, d_out);
}
